// MultiHeadSelfAttention_14328010899543
// MI455X (gfx1250) — matrix operand order repaired and checked by reading, not compiled, not run on hardware
//
#include <hip/hip_runtime.h>
#include <cstdint>

// ---------------------------------------------------------------------------
// Problem constants (from reference): B=2, N=2048, D=1024, H=16, S=64
// ---------------------------------------------------------------------------
constexpr int Bc   = 2;
constexpr int Nc   = 2048;
constexpr int Dc   = 1024;
constexpr int Hc   = 16;
constexpr int Sc   = 64;
constexpr int Mc   = Bc * Nc;      // 4096 token rows
constexpr int BHc  = Bc * Hc;      // 32
constexpr int N3D  = 3 * Dc;       // 3072

typedef __bf16 bf16;
typedef __attribute__((ext_vector_type(16))) bf16  v16bf;
typedef __attribute__((ext_vector_type(8)))  bf16  v8bf;
typedef __attribute__((ext_vector_type(4)))  bf16  v4bf;
typedef __attribute__((ext_vector_type(8)))  float v8f;
typedef __attribute__((ext_vector_type(4)))  float v4f;
typedef __attribute__((ext_vector_type(4)))  unsigned int u32x4;
typedef __attribute__((ext_vector_type(8)))  int i32x8;
typedef __attribute__((ext_vector_type(4)))  int i32x4;

#if defined(__has_builtin)
#if __has_builtin(__builtin_amdgcn_tensor_load_to_lds) && \
    __has_builtin(__builtin_amdgcn_s_wait_tensorcnt)
#define HAVE_TDM 1
#endif
#endif
#ifndef HAVE_TDM
#define HAVE_TDM 0
#endif

static __device__ inline v16bf cat8(v8bf a, v8bf b) {
  return __builtin_shufflevector(a, b, 0,1,2,3,4,5,6,7,8,9,10,11,12,13,14,15);
}

static __device__ inline v8f wmma_bf16(v16bf a, v16bf b, v8f c) {
  return __builtin_amdgcn_wmma_f32_16x16x32_bf16(
      false, a, false, b, (short)0, c, false, false);
}

static __device__ inline float rmax16(float v) {
  v = fmaxf(v, __shfl_xor(v, 1));
  v = fmaxf(v, __shfl_xor(v, 2));
  v = fmaxf(v, __shfl_xor(v, 4));
  v = fmaxf(v, __shfl_xor(v, 8));
  return v;
}
static __device__ inline float rsum16(float v) {
  v += __shfl_xor(v, 1);
  v += __shfl_xor(v, 2);
  v += __shfl_xor(v, 4);
  v += __shfl_xor(v, 8);
  return v;
}

#if HAVE_TDM
// ---------------------------------------------------------------------------
// Tensor Data Mover: 2-D tile load Global -> LDS.
// D# group0/group1 packed per CDNA5 ISA 8.3/8.4; groups 2/3 zero (<=2D).
// dims/strides in data_size units; ds_log2: 0=1B, 1=2B.
// ---------------------------------------------------------------------------
static __device__ inline void tdm_load_2d(void* lds_ptr, const void* gptr,
                                          int ds_log2,
                                          unsigned tile0, unsigned tile1,
                                          unsigned long long stride0) {
  const unsigned long long ga = (unsigned long long)(uintptr_t)gptr;
  const unsigned lds = (unsigned)(uintptr_t)lds_ptr;  // low 32 bits = LDS offset
  u32x4 g0;
  g0[0] = 1u;                                          // count=1, user mode
  g0[1] = lds;                                         // lds_addr [63:32]
  g0[2] = (unsigned)(ga & 0xFFFFFFFFu);                // global_addr [95:64]
  g0[3] = (unsigned)((ga >> 32) & 0x01FFFFFFu) | (2u << 30);  // addr hi | type=2
  i32x8 g1;
  g1[0] = (int)((unsigned)(ds_log2 & 3) << 16);        // wg_mask=0, data_size
  g1[1] = (int)((tile0 & 0xFFFFu) << 16);              // tensor_dim0[15:0]
  g1[2] = (int)(((tile0 >> 16) & 0xFFFFu) |            // tensor_dim0[31:16]
                ((tile1 & 0xFFFFu) << 16));            // tensor_dim1[15:0]
  g1[3] = (int)(((tile1 >> 16) & 0xFFFFu) |            // tensor_dim1[31:16]
                ((tile0 & 0xFFFFu) << 16));            // tile_dim0
  g1[4] = (int)(tile1 & 0xFFFFu);                      // tile_dim1 (tile_dim2=0)
  g1[5] = (int)(unsigned)(stride0 & 0xFFFFFFFFull);    // tensor_dim0_stride lo
  g1[6] = (int)(unsigned)((stride0 >> 32) & 0xFFFFull);// stride hi (dim1_stride=0)
  g1[7] = 0;
  i32x4 z4 = {0, 0, 0, 0};
#if __clang_major__ >= 23
  i32x8 z8 = {0, 0, 0, 0, 0, 0, 0, 0};
  __builtin_amdgcn_tensor_load_to_lds(g0, g1, z4, z4, z8, 0);
#else
  __builtin_amdgcn_tensor_load_to_lds(g0, g1, z4, z4, 0);
#endif
}
#endif  // HAVE_TDM

// ---------------------------------------------------------------------------
// fp32 -> bf16 bulk convert (one-time; removes cvt from GEMM hot loops)
// ---------------------------------------------------------------------------
__global__ __launch_bounds__(256) void cvt_f32_to_bf16(
    const float* __restrict__ s, bf16* __restrict__ d, int n4) {
  const int i = blockIdx.x * 256 + threadIdx.x;
  if (i < n4) {
    v4f a = *(const v4f*)(s + (size_t)i * 4);
    v4bf b = {(bf16)a.x, (bf16)a.y, (bf16)a.z, (bf16)a.w};
    *(v4bf*)(d + (size_t)i * 4) = b;
  }
}

// ---------------------------------------------------------------------------
// WMMA GEMM:  C[M,Nout] = A[M,K](bf16) * W[Nout,K]^T(bf16) + bias(f32)
// MODE 0: scatter to bf16 Q[bh][n][s], K[bh][n][s], Vt[bh][s][n]
// MODE 1: write fp32 row-major out[M,Nout]
// 256 thr = 8 waves; tile 128x128; K-step 32; TDM double-buffered staging.
// ---------------------------------------------------------------------------
template <int MODE>
__global__ __launch_bounds__(256) void gemm_wmma(
    const bf16* __restrict__ A, const bf16* __restrict__ W,
    const float* __restrict__ bias, int K, int Nout,
    float* __restrict__ outF,
    bf16* __restrict__ qb, bf16* __restrict__ kb, bf16* __restrict__ vb) {
  __shared__ bf16 As[2][128 * 32];
  __shared__ bf16 Bs[2][128 * 32];

  const int tid  = threadIdx.x;
  const int lane = tid & 31;
  const int wid  = tid >> 5;
  const int wm   = wid >> 1;        // 0..3 (M)
  const int wn   = wid & 1;         // 0..1 (N)
  const int lo   = lane & 15;
  const int hi   = lane >> 4;
  const int m0   = blockIdx.x * 128;
  const int n0   = blockIdx.y * 128;
  const int KSTEPS = K / 32;

  v8f acc[2][4];
#pragma unroll
  for (int fm = 0; fm < 2; ++fm)
#pragma unroll
    for (int fn = 0; fn < 4; ++fn)
      acc[fm][fn] = v8f{0.f, 0.f, 0.f, 0.f, 0.f, 0.f, 0.f, 0.f};

#if HAVE_TDM
  if (wid == 0) {  // prologue: stage K-step 0 into buffer 0
    tdm_load_2d(&As[0][0], A + (size_t)m0 * K, 1, 32, 128, (unsigned long long)K);
    tdm_load_2d(&Bs[0][0], W + (size_t)n0 * K, 1, 32, 128, (unsigned long long)K);
  }
#endif

  for (int it = 0; it < KSTEPS; ++it) {
#if HAVE_TDM
    const int cur = it & 1;
    if (wid == 0) {
      if (it + 1 < KSTEPS) {  // prefetch next K-step into alternate buffer
        const int nk = (it + 1) * 32;
        tdm_load_2d(&As[cur ^ 1][0], A + (size_t)m0 * K + nk, 1, 32, 128,
                    (unsigned long long)K);
        tdm_load_2d(&Bs[cur ^ 1][0], W + (size_t)n0 * K + nk, 1, 32, 128,
                    (unsigned long long)K);
        __builtin_amdgcn_s_wait_tensorcnt((short)2);  // current buffer done
      } else {
        __builtin_amdgcn_s_wait_tensorcnt((short)0);
      }
    }
    __syncthreads();
#else
    const int cur = 0;
    const int k0 = it * 32;
    __syncthreads();
#pragma unroll
    for (int t = 0; t < 2; ++t) {
      const int u = tid + 256 * t;       // 0..511
      const int row = u >> 2;            // 0..127
      const int seg = (u & 3) * 8;
      *(v8bf*)(&As[0][row * 32 + seg]) =
          *(const v8bf*)(A + (size_t)(m0 + row) * K + k0 + seg);
      *(v8bf*)(&Bs[0][row * 32 + seg]) =
          *(const v8bf*)(W + (size_t)(n0 + row) * K + k0 + seg);
    }
    __syncthreads();
#endif

    v16bf af[2];
#pragma unroll
    for (int fm = 0; fm < 2; ++fm) {
      const int row = wm * 32 + fm * 16 + lo;
      v8bf a0 = *(const v8bf*)(&As[cur][row * 32 + hi * 8]);
      v8bf a1 = *(const v8bf*)(&As[cur][row * 32 + hi * 8 + 16]);
      af[fm] = cat8(a0, a1);
    }
    v16bf bfv[4];
#pragma unroll
    for (int fn = 0; fn < 4; ++fn) {
      const int row = wn * 64 + fn * 16 + lo;
      v8bf b0 = *(const v8bf*)(&Bs[cur][row * 32 + hi * 8]);
      v8bf b1 = *(const v8bf*)(&Bs[cur][row * 32 + hi * 8 + 16]);
      bfv[fn] = cat8(b0, b1);
    }
#pragma unroll
    for (int fm = 0; fm < 2; ++fm)
#pragma unroll
      for (int fn = 0; fn < 4; ++fn)
        acc[fm][fn] = wmma_bf16(af[fm], bfv[fn], acc[fm][fn]);
    __syncthreads();
  }

  // epilogue: element r -> row = base + r + 8*hi, col = base + lo
#pragma unroll
  for (int fn = 0; fn < 4; ++fn) {
    const int j = n0 + wn * 64 + fn * 16 + lo;
    const float bv = bias[j];
#pragma unroll
    for (int fm = 0; fm < 2; ++fm) {
#pragma unroll
      for (int r = 0; r < 8; ++r) {
        const int mrow = m0 + wm * 32 + fm * 16 + r + 8 * hi;
        const float val = acc[fm][fn][r] + bv;
        if (MODE == 1) {
          outF[(size_t)mrow * Nout + j] = val;
        } else {
          const int which = j >> 10;       // 0=Q,1=K,2=V
          const int dcol  = j & (Dc - 1);
          const int h     = dcol >> 6;
          const int s     = dcol & (Sc - 1);
          const int b     = mrow >> 11;
          const int n     = mrow & (Nc - 1);
          const int bh    = b * Hc + h;
          const bf16 v    = (bf16)val;
          if (which == 0)      qb[((size_t)bh * Nc + n) * Sc + s] = v;
          else if (which == 1) kb[((size_t)bh * Nc + n) * Sc + s] = v;
          else                 vb[((size_t)bh * Sc + s) * Nc + n] = v;  // V^T
        }
      }
    }
  }
}

// ---------------------------------------------------------------------------
// Flash attention: block = 64 query rows of one (b,h); 128 thr = 4 waves.
// TDM stages K/V^T/mask tiles; WMMA for QK^T and PV; online softmax in fp32.
// Output bf16, token-major [b*N+n][h*S+s] (feeds the out-projection GEMM).
// ---------------------------------------------------------------------------
__global__ __launch_bounds__(128) void flash_attn(
    const bf16* __restrict__ qb, const bf16* __restrict__ kb,
    const bf16* __restrict__ vb, const unsigned char* __restrict__ mask,
    bf16* __restrict__ attn) {
  __shared__ bf16 kS[64 * 64];              // K tile [j][s]
  __shared__ bf16 vS[64 * 64];              // V^T tile [s][j]
  __shared__ unsigned char mS[64 * 64];     // mask tile [q][j]
  __shared__ bf16 pS[4][16 * 64];           // per-wave P tile [m][j]

  const int tid  = threadIdx.x;
  const int lane = tid & 31;
  const int wid  = tid >> 5;
  const int lo   = lane & 15;
  const int hi   = lane >> 4;
  const int bh   = blockIdx.y;
  const int b    = bh >> 4;
  const int h    = bh & (Hc - 1);
  const int q0   = blockIdx.x * 64;
  const float scale = 0.125f;               // 1/sqrt(64)
  const float BIG_NEG = -3.0e38f;

  // Q fragments (A layout) for this wave's 16 rows, 2 K-steps over S=64
  const bf16* qbase = qb + ((size_t)bh * Nc + q0 + wid * 16 + lo) * Sc;
  v16bf qf[2];
#pragma unroll
  for (int kk = 0; kk < 2; ++kk) {
    v8bf a0 = *(const v8bf*)(qbase + kk * 32 + hi * 8);
    v8bf a1 = *(const v8bf*)(qbase + kk * 32 + hi * 8 + 16);
    qf[kk] = cat8(a0, a1);
  }

  v8f o[4];
#pragma unroll
  for (int oc = 0; oc < 4; ++oc)
    o[oc] = v8f{0.f, 0.f, 0.f, 0.f, 0.f, 0.f, 0.f, 0.f};
  float mrow[8], lrow[8];
#pragma unroll
  for (int r = 0; r < 8; ++r) { mrow[r] = BIG_NEG; lrow[r] = 0.f; }

  for (int j0 = 0; j0 < Nc; j0 += 64) {
    __syncthreads();   // all waves done reading previous tiles
#if HAVE_TDM
    if (wid == 0) {
      tdm_load_2d(kS, kb + ((size_t)bh * Nc + j0) * Sc, 1, 64, 64, 64ull);
      tdm_load_2d(vS, vb + (size_t)bh * Sc * Nc + j0, 1, 64, 64,
                  (unsigned long long)Nc);
      tdm_load_2d(mS, mask + (size_t)b * Nc * Nc + (size_t)q0 * Nc + j0, 0,
                  64, 64, (unsigned long long)Nc);
      __builtin_amdgcn_s_wait_tensorcnt((short)0);
    }
#else
#pragma unroll
    for (int i = 0; i < 4; ++i) {
      const int u = tid + 128 * i;
      const int row = u >> 3;
      const int seg = (u & 7) * 8;
      *(v8bf*)(&kS[row * 64 + seg]) =
          *(const v8bf*)(kb + ((size_t)bh * Nc + j0 + row) * Sc + seg);
      *(v8bf*)(&vS[row * 64 + seg]) =
          *(const v8bf*)(vb + ((size_t)bh * Sc + row) * Nc + j0 + seg);
    }
    {
      const int r = tid >> 1;
      const int c = (tid & 1) * 32;
      const unsigned char* mg =
          mask + (size_t)b * Nc * Nc + (size_t)(q0 + r) * Nc + j0 + c;
      *(uint4*)(&mS[r * 64 + c])      = *(const uint4*)(mg);
      *(uint4*)(&mS[r * 64 + c + 16]) = *(const uint4*)(mg + 16);
    }
#endif
    __syncthreads();

    // --- scores: 16x64 per wave via WMMA ---
    v8f sc[4];
#pragma unroll
    for (int jc = 0; jc < 4; ++jc) {
      const int krow = jc * 16 + lo;
      v8f c = v8f{0.f, 0.f, 0.f, 0.f, 0.f, 0.f, 0.f, 0.f};
#pragma unroll
      for (int kk = 0; kk < 2; ++kk) {
        v8bf b0 = *(const v8bf*)(&kS[krow * 64 + kk * 32 + hi * 8]);
        v8bf b1 = *(const v8bf*)(&kS[krow * 64 + kk * 32 + hi * 8 + 16]);
        c = wmma_bf16(qf[kk], cat8(b0, b1), c);
      }
      sc[jc] = c;
    }
    // --- scale + mask ---
#pragma unroll
    for (int jc = 0; jc < 4; ++jc)
#pragma unroll
      for (int r = 0; r < 8; ++r) {
        float s = sc[jc][r] * scale;
        if (mS[(wid * 16 + r + 8 * hi) * 64 + jc * 16 + lo]) s = BIG_NEG;
        sc[jc][r] = s;
      }
    // --- online softmax ---
    float fac[8];
#pragma unroll
    for (int r = 0; r < 8; ++r) {
      float v = fmaxf(fmaxf(sc[0][r], sc[1][r]), fmaxf(sc[2][r], sc[3][r]));
      v = rmax16(v);
      const float newm = fmaxf(mrow[r], v);
      fac[r] = __expf(mrow[r] - newm);
      mrow[r] = newm;
    }
#pragma unroll
    for (int jc = 0; jc < 4; ++jc)
#pragma unroll
      for (int r = 0; r < 8; ++r) sc[jc][r] = __expf(sc[jc][r] - mrow[r]);
#pragma unroll
    for (int r = 0; r < 8; ++r) {
      float s = sc[0][r] + sc[1][r] + sc[2][r] + sc[3][r];
      s = rsum16(s);
      lrow[r] = lrow[r] * fac[r] + s;
    }
#pragma unroll
    for (int oc = 0; oc < 4; ++oc)
#pragma unroll
      for (int r = 0; r < 8; ++r) o[oc][r] *= fac[r];

    // --- P: C-layout regs -> per-wave LDS -> A-layout frags ---
    bf16* pw = pS[wid];
#pragma unroll
    for (int jc = 0; jc < 4; ++jc)
#pragma unroll
      for (int r = 0; r < 8; ++r)
        pw[(r + 8 * hi) * 64 + jc * 16 + lo] = (bf16)sc[jc][r];
    __syncthreads();
    v16bf pf[2];
#pragma unroll
    for (int kk = 0; kk < 2; ++kk) {
      v8bf p0 = *(const v8bf*)(&pw[lo * 64 + kk * 32 + hi * 8]);
      v8bf p1 = *(const v8bf*)(&pw[lo * 64 + kk * 32 + hi * 8 + 16]);
      pf[kk] = cat8(p0, p1);
    }
    // --- O += P @ V ---
#pragma unroll
    for (int oc = 0; oc < 4; ++oc) {
      const int vrow = oc * 16 + lo;
#pragma unroll
      for (int kk = 0; kk < 2; ++kk) {
        v8bf b0 = *(const v8bf*)(&vS[vrow * 64 + kk * 32 + hi * 8]);
        v8bf b1 = *(const v8bf*)(&vS[vrow * 64 + kk * 32 + hi * 8 + 16]);
        o[oc] = wmma_bf16(pf[kk], cat8(b0, b1), o[oc]);
      }
    }
  }

  // --- normalize + write bf16 token-major output ---
#pragma unroll
  for (int oc = 0; oc < 4; ++oc) {
#pragma unroll
    for (int r = 0; r < 8; ++r) {
      const int qrow = q0 + wid * 16 + r + 8 * hi;
      const float val = o[oc][r] / lrow[r];
      attn[((size_t)(b * Nc + qrow)) * Dc + h * Sc + oc * 16 + lo] = (bf16)val;
    }
  }
}

// ---------------------------------------------------------------------------
// Launch: cvt x/Wqkv/Wout -> qkv GEMM -> flash attention -> out GEMM
// ---------------------------------------------------------------------------
extern "C" void kernel_launch(void* const* d_in, const int* in_sizes, int n_in,
                              void* d_out, int out_size, void* d_ws,
                              size_t ws_size, hipStream_t stream) {
  (void)in_sizes; (void)n_in; (void)out_size; (void)ws_size;
  const float* x    = (const float*)d_in[0];
  const unsigned char* mask = (const unsigned char*)d_in[1];  // bool [B,N,N]
  const float* Wqkv = (const float*)d_in[2];
  const float* bqkv = (const float*)d_in[3];
  const float* Wout = (const float*)d_in[4];
  const float* bout = (const float*)d_in[5];
  float* out = (float*)d_out;

  const size_t xN    = (size_t)Mc * Dc;        // 4,194,304
  const size_t wqkvN = (size_t)N3D * Dc;       // 3,145,728
  const size_t woutN = (size_t)Dc * Dc;        // 1,048,576
  const size_t headN = (size_t)BHc * Nc * Sc;  // 4,194,304

  bf16* xb    = (bf16*)d_ws;
  bf16* wqkvb = xb + xN;
  bf16* woutb = wqkvb + wqkvN;
  bf16* qb    = woutb + woutN;
  bf16* kb    = qb + headN;
  bf16* vb    = kb + headN;
  bf16* attnb = vb + headN;                    // [M, D] bf16

  // 0) one-time fp32 -> bf16 conversions
  cvt_f32_to_bf16<<<(int)(xN / 4 / 256),    256, 0, stream>>>(x,    xb,    (int)(xN / 4));
  cvt_f32_to_bf16<<<(int)(wqkvN / 4 / 256), 256, 0, stream>>>(Wqkv, wqkvb, (int)(wqkvN / 4));
  cvt_f32_to_bf16<<<(int)(woutN / 4 / 256), 256, 0, stream>>>(Wout, woutb, (int)(woutN / 4));

  // 1) QKV projection -> head-major Q/K and transposed V (bf16)
  dim3 g1(Mc / 128, N3D / 128);
  gemm_wmma<0><<<g1, 256, 0, stream>>>(xb, wqkvb, bqkv, Dc, N3D,
                                       nullptr, qb, kb, vb);
  // 2) Flash attention
  dim3 g2(Nc / 64, BHc);
  flash_attn<<<g2, 128, 0, stream>>>(qb, kb, vb, mask, attnb);
  // 3) Output projection -> fp32 result
  dim3 g3(Mc / 128, Dc / 128);
  gemm_wmma<1><<<g3, 256, 0, stream>>>(attnb, woutb, bout, Dc, Dc,
                                       out, nullptr, nullptr, nullptr);
}
